// _MultiHeadAttnLogitsFusion_27642409517738
// MI455X (gfx1250) — compile-verified, matrix operand order checked by reading, not run on hardware
//
#include <hip/hip_runtime.h>
#include <math.h>

// ---------------- problem constants ----------------
#define B_ 16
#define L_ 512
#define D_ 768
#define H_ 12
#define I_ 3072
#define DH_ 64
#define SCALE_ 0.125f
#define EPS_ 1e-12f

typedef __attribute__((ext_vector_type(16))) __bf16 v16bf;
typedef __attribute__((ext_vector_type(8)))  __bf16 v8bf;
typedef __attribute__((ext_vector_type(8)))  float  v8f;

// native f32 -> bf16 (RNE via hardware convert)
__device__ __forceinline__ __bf16 f2bf(float x) { return (__bf16)x; }
__device__ __forceinline__ unsigned pack2bf(float a, float b) {
  union { __bf16 h[2]; unsigned u; } o;
  o.h[0] = (__bf16)a; o.h[1] = (__bf16)b;
  return o.u;
}

// Fragment loaders. CDNA5 16-bit A/B layout: lane (idx,hi) holds
// K = {kb + 8*hi + e} (e<8) and K = {kb + 16 + 8*hi + e} (upper 8).
// base is wave-uniform; off is a 32-bit lane-varying element offset so the
// backend can select global_load_b128 (saddr + voffset) addressing.
__device__ __forceinline__ v16bf frag_ld_bf(const __bf16* __restrict__ base, unsigned off) {
  v8bf lo = *(const v8bf*)(base + (size_t)off);
  v8bf hi = *(const v8bf*)(base + (size_t)off + 16);
  return __builtin_shufflevector(lo, hi, 0, 1, 2, 3, 4, 5, 6, 7,
                                 8, 9, 10, 11, 12, 13, 14, 15);
}
__device__ __forceinline__ v16bf frag_ld_g(const float* __restrict__ base, unsigned off) {
  const float* p = base + (size_t)off;
  float4 a = *(const float4*)(p);
  float4 b = *(const float4*)(p + 4);
  float4 c = *(const float4*)(p + 16);
  float4 d = *(const float4*)(p + 20);
  v16bf r;
  r[0] = f2bf(a.x); r[1] = f2bf(a.y); r[2]  = f2bf(a.z); r[3]  = f2bf(a.w);
  r[4] = f2bf(b.x); r[5] = f2bf(b.y); r[6]  = f2bf(b.z); r[7]  = f2bf(b.w);
  r[8] = f2bf(c.x); r[9] = f2bf(c.y); r[10] = f2bf(c.z); r[11] = f2bf(c.w);
  r[12] = f2bf(d.x); r[13] = f2bf(d.y); r[14] = f2bf(d.z); r[15] = f2bf(d.w);
  return r;
}
__device__ __forceinline__ v16bf frag_ld_lds(const __bf16* p) {
  v16bf r;
#pragma unroll
  for (int e = 0; e < 8; ++e) { r[e] = p[e]; r[e + 8] = p[16 + e]; }
  return r;
}
// B row-major [K,N] bf16 in global (strided over K)
__device__ __forceinline__ v16bf frag_ld_bf_strided(const __bf16* __restrict__ base,
                                                    unsigned off, int stride) {
  v16bf r;
#pragma unroll
  for (int e = 0; e < 8; ++e) {
    r[e]     = base[(size_t)(off + (unsigned)(e * stride))];
    r[e + 8] = base[(size_t)(off + (unsigned)((e + 16) * stride))];
  }
  return r;
}

__device__ __forceinline__ float gelu_exact(float v) {
  return 0.5f * v * (1.0f + erff(v * 0.70710678118654752f));
}

// =====================================================================
// Generic GEMM: C[M,N] = act(A[M,K] @ W[K,N] + bias)
// Block tile 128x128, 8 waves (4 M x 2 N), wave tile 32x64 -> 2x4 WMMAs.
// OUTBF: 1 -> write bf16 (for WMMA consumers), 0 -> write f32.
// =====================================================================
#define GM 128
#define GN 128
#define GK 32
template <int ACT, int OUTBF>
__global__ __launch_bounds__(256) void gemm_bf16_wmma(
    const float* __restrict__ A, const float* __restrict__ W,
    const float* __restrict__ bias, void* __restrict__ Cv,
    int M, int N, int K) {
  __shared__ __bf16 sA[GM * GK];
  __shared__ __bf16 sB[GN * GK];  // transposed: [n][k]
  const int t = threadIdx.x;
  const int wid = t >> 5, lane = t & 31;
  const int idx = lane & 15, hi = lane >> 4;
  const int wm = wid & 3, wn = wid >> 2;
  const int gm0 = blockIdx.y * GM, gn0 = blockIdx.x * GN;

  v8f acc[2][4];
#pragma unroll
  for (int i = 0; i < 2; ++i)
#pragma unroll
    for (int j = 0; j < 4; ++j)
#pragma unroll
      for (int r = 0; r < 8; ++r) acc[i][j][r] = 0.0f;

  for (int kt = 0; kt < K; kt += GK) {
    // A tile: 128x32 f32 -> bf16, float4 global loads, packed b32 LDS stores
#pragma unroll
    for (int i = 0; i < (GM * GK) / (256 * 4); ++i) {
      int e = (i * 256 + t) * 4; int r = e >> 5, c = e & 31;
      float4 v = *(const float4*)&A[(size_t)(gm0 + r) * K + kt + c];
      unsigned* dst = (unsigned*)&sA[r * GK + c];
      dst[0] = pack2bf(v.x, v.y);
      dst[1] = pack2bf(v.z, v.w);
      if (kt + GK < K)  // prefetch next K tile (global_prefetch_b8)
        __builtin_prefetch(&A[(size_t)(gm0 + r) * K + kt + GK + c], 0, 3);
    }
    // W tile: 32x128 f32, float4 loads along n, transposed b16 LDS stores
#pragma unroll
    for (int i = 0; i < (GN * GK) / (256 * 4); ++i) {
      int e = (i * 256 + t) * 4; int r = e >> 7, c = e & 127;
      float4 v = *(const float4*)&W[(size_t)(kt + r) * N + gn0 + c];
      sB[(c + 0) * GK + r] = f2bf(v.x);
      sB[(c + 1) * GK + r] = f2bf(v.y);
      sB[(c + 2) * GK + r] = f2bf(v.z);
      sB[(c + 3) * GK + r] = f2bf(v.w);
      if (kt + GK < K)
        __builtin_prefetch(&W[(size_t)(kt + GK + r) * N + gn0 + c], 0, 3);
    }
    __syncthreads();
    v16bf af[2], bfar[4];
#pragma unroll
    for (int i = 0; i < 2; ++i)
      af[i] = frag_ld_lds(&sA[(wm * 32 + i * 16 + idx) * GK + 8 * hi]);
#pragma unroll
    for (int j = 0; j < 4; ++j)
      bfar[j] = frag_ld_lds(&sB[(wn * 64 + j * 16 + idx) * GK + 8 * hi]);
#pragma unroll
    for (int i = 0; i < 2; ++i)
#pragma unroll
      for (int j = 0; j < 4; ++j)
        acc[i][j] = __builtin_amdgcn_wmma_f32_16x16x32_bf16(
            false, af[i], false, bfar[j], (short)0, acc[i][j], false, false);
    __syncthreads();
  }
#pragma unroll
  for (int i = 0; i < 2; ++i)
#pragma unroll
    for (int j = 0; j < 4; ++j) {
      int col = gn0 + wn * 64 + j * 16 + idx;
      float bv = bias[col];
#pragma unroll
      for (int r = 0; r < 8; ++r) {
        int row = gm0 + wm * 32 + i * 16 + r + 8 * hi;
        float v = acc[i][j][r] + bv;
        if (ACT == 1) v = gelu_exact(v);
        if (OUTBF)
          ((__bf16*)Cv)[(size_t)row * N + col] = (__bf16)v;
        else
          ((float*)Cv)[(size_t)row * N + col] = v;
      }
    }
}

// =====================================================================
// Fused gated attention logits: out[bh,q,k] = sum_s gains[s] * Qs Ks^T
// grid (L/64, L/128, B*H). Q/K bf16 [B,L,D]; b128 fragment loads, no
// conversion VALU. Output logits f32.
// =====================================================================
#define AM 128
#define AN 64
__global__ __launch_bounds__(256) void attn_logits_wmma(
    const __bf16* __restrict__ q0, const __bf16* __restrict__ k0,
    const __bf16* __restrict__ q1, const __bf16* __restrict__ k1,
    const __bf16* __restrict__ q2, const __bf16* __restrict__ k2,
    const float* __restrict__ gains, int nsrc, float* __restrict__ out) {
  const int z = blockIdx.z, b = z / H_, h = z % H_;
  const int t = threadIdx.x;
  const int wid = t >> 5, lane = t & 31;
  const int idx = lane & 15, hi = lane >> 4;
  const int wm = wid & 3, wn = wid >> 2;
  const int gm0 = blockIdx.y * AM, gn0 = blockIdx.x * AN;
  const __bf16* qs[3] = {q0, q1, q2};
  const __bf16* ks[3] = {k0, k1, k2};
  const unsigned head = (unsigned)(b * L_) * D_ + (unsigned)(h * DH_);

  v8f acc[2][2];
#pragma unroll
  for (int i = 0; i < 2; ++i)
#pragma unroll
    for (int j = 0; j < 2; ++j)
#pragma unroll
      for (int r = 0; r < 8; ++r) acc[i][j][r] = 0.0f;

  for (int s = 0; s < nsrc; ++s) {
    const __bf16* qb = qs[s];
    const __bf16* kb = ks[s];
    v8f part[2][2];
#pragma unroll
    for (int i = 0; i < 2; ++i)
#pragma unroll
      for (int j = 0; j < 2; ++j)
#pragma unroll
        for (int r = 0; r < 8; ++r) part[i][j][r] = 0.0f;
#pragma unroll
    for (int kt = 0; kt < DH_; kt += 32) {
      v16bf af[2], bfar[2];
#pragma unroll
      for (int i = 0; i < 2; ++i)
        af[i] = frag_ld_bf(qb, head + (unsigned)(gm0 + wm * 32 + i * 16 + idx) * D_
                                    + (unsigned)(kt + 8 * hi));
#pragma unroll
      for (int j = 0; j < 2; ++j)
        bfar[j] = frag_ld_bf(kb, head + (unsigned)(gn0 + wn * 32 + j * 16 + idx) * D_
                                      + (unsigned)(kt + 8 * hi));
#pragma unroll
      for (int i = 0; i < 2; ++i)
#pragma unroll
        for (int j = 0; j < 2; ++j)
          part[i][j] = __builtin_amdgcn_wmma_f32_16x16x32_bf16(
              false, af[i], false, bfar[j], (short)0, part[i][j], false, false);
    }
    float g = gains[s];
#pragma unroll
    for (int i = 0; i < 2; ++i)
#pragma unroll
      for (int j = 0; j < 2; ++j)
#pragma unroll
        for (int r = 0; r < 8; ++r) acc[i][j][r] += g * part[i][j][r];
  }
#pragma unroll
  for (int i = 0; i < 2; ++i)
#pragma unroll
    for (int j = 0; j < 2; ++j) {
      unsigned col = (unsigned)(gn0 + wn * 32 + j * 16 + idx);
#pragma unroll
      for (int r = 0; r < 8; ++r) {
        unsigned row = (unsigned)(gm0 + wm * 32 + i * 16 + r + 8 * hi);
        out[(size_t)(((unsigned)z * L_ + row) * L_ + col)] = acc[i][j][r];
      }
    }
}

// =====================================================================
// ctx[b,q,h*DH+n] = sum_k P[bh,q,k] * V[b,k,h*DH+n]   grid (1, L/128, B*H)
// P f32 (converted once here), V bf16. Output f32 (feeds o-proj GEMM).
// =====================================================================
__global__ __launch_bounds__(256) void attn_ctx_wmma(
    const float* __restrict__ attn, const __bf16* __restrict__ vv,
    float* __restrict__ ctx) {
  const int z = blockIdx.z, b = z / H_, h = z % H_;
  const int t = threadIdx.x;
  const int wid = t >> 5, lane = t & 31;
  const int idx = lane & 15, hi = lane >> 4;
  const int wm = wid & 3, wn = wid >> 2;
  const int gm0 = blockIdx.y * AM;
  const unsigned abase = (unsigned)z * (unsigned)(L_ * L_);
  const unsigned vbase = (unsigned)(b * L_) * D_ + (unsigned)(h * DH_);

  v8f acc[2][2];
#pragma unroll
  for (int i = 0; i < 2; ++i)
#pragma unroll
    for (int j = 0; j < 2; ++j)
#pragma unroll
      for (int r = 0; r < 8; ++r) acc[i][j][r] = 0.0f;

  for (int kt = 0; kt < L_; kt += 32) {
    v16bf af[2], bfar[2];
#pragma unroll
    for (int i = 0; i < 2; ++i)
      af[i] = frag_ld_g(attn, abase + (unsigned)(gm0 + wm * 32 + i * 16 + idx) * L_
                                    + (unsigned)(kt + 8 * hi));
#pragma unroll
    for (int j = 0; j < 2; ++j)
      bfar[j] = frag_ld_bf_strided(
          vv, vbase + (unsigned)(kt + 8 * hi) * D_ + (unsigned)(wn * 32 + j * 16 + idx), D_);
#pragma unroll
    for (int i = 0; i < 2; ++i)
#pragma unroll
      for (int j = 0; j < 2; ++j)
        acc[i][j] = __builtin_amdgcn_wmma_f32_16x16x32_bf16(
            false, af[i], false, bfar[j], (short)0, acc[i][j], false, false);
  }
#pragma unroll
  for (int i = 0; i < 2; ++i)
#pragma unroll
    for (int j = 0; j < 2; ++j) {
      unsigned col = (unsigned)(wn * 32 + j * 16 + idx);
#pragma unroll
      for (int r = 0; r < 8; ++r) {
        unsigned row = (unsigned)(gm0 + wm * 32 + i * 16 + r + 8 * hi);
        ctx[(size_t)(((unsigned)(b * L_) + row) * D_ + (unsigned)(h * DH_) + col)] = acc[i][j][r];
      }
    }
}

// =====================================================================
// In-place row softmax over L with x = x*scale + mask[q,:]. Wave per row.
// =====================================================================
__global__ __launch_bounds__(256) void softmax_rows(
    float* __restrict__ x, const float* __restrict__ mask, float scale) {
  const int wid = threadIdx.x >> 5, lane = threadIdx.x & 31;
  const size_t row = (size_t)blockIdx.x * 8 + wid;  // over B*H*L
  const int q = (int)(row % L_);
  float* p = x + row * L_;
  const float* mp = mask + (size_t)q * L_;
  float vals[L_ / 32];
  float m = -3.4e38f;
#pragma unroll
  for (int i = 0; i < L_ / 32; ++i) {
    int c = lane + 32 * i;
    vals[i] = p[c] * scale + mp[c];
    m = fmaxf(m, vals[i]);
  }
#pragma unroll
  for (int o = 16; o > 0; o >>= 1) m = fmaxf(m, __shfl_xor(m, o, 32));
  float s = 0.0f;
#pragma unroll
  for (int i = 0; i < L_ / 32; ++i) { vals[i] = expf(vals[i] - m); s += vals[i]; }
#pragma unroll
  for (int o = 16; o > 0; o >>= 1) s += __shfl_xor(s, o, 32);
  float inv = 1.0f / s;
#pragma unroll
  for (int i = 0; i < L_ / 32; ++i) p[lane + 32 * i] = vals[i] * inv;
}

// =====================================================================
// out[row] = LayerNorm(x[row] + res[row]; g, b), rows of width D. Wave/row.
// =====================================================================
__global__ __launch_bounds__(256) void add_ln_rows(
    const float* __restrict__ x, const float* __restrict__ res,
    const float* __restrict__ g, const float* __restrict__ bta,
    float* __restrict__ out) {
  const int wid = threadIdx.x >> 5, lane = threadIdx.x & 31;
  const size_t row = (size_t)blockIdx.x * 8 + wid;  // over B*L
  const float* px = x + row * D_;
  const float* pr = res + row * D_;
  float v[D_ / 32];
  float s = 0.0f;
#pragma unroll
  for (int i = 0; i < D_ / 32; ++i) {
    v[i] = px[lane + 32 * i] + pr[lane + 32 * i];
    s += v[i];
  }
#pragma unroll
  for (int o = 16; o > 0; o >>= 1) s += __shfl_xor(s, o, 32);
  float mean = s * (1.0f / D_);
  float var = 0.0f;
#pragma unroll
  for (int i = 0; i < D_ / 32; ++i) { float d = v[i] - mean; var += d * d; }
#pragma unroll
  for (int o = 16; o > 0; o >>= 1) var += __shfl_xor(var, o, 32);
  var *= (1.0f / D_);
  float inv = rsqrtf(var + EPS_);
#pragma unroll
  for (int i = 0; i < D_ / 32; ++i) {
    int c = lane + 32 * i;
    out[row * D_ + c] = (v[i] - mean) * inv * g[c] + bta[c];
  }
}

__global__ void setup_gains(const float* __restrict__ gate, float* __restrict__ G) {
  int t = threadIdx.x;
  if (t < 3) G[t] = 1.0f / (1.0f + expf(-gate[t]));
  if (t == 3) G[3] = 1.0f;
}

// =====================================================================
extern "C" void kernel_launch(void* const* d_in, const int* in_sizes, int n_in,
                              void* d_out, int out_size, void* d_ws, size_t ws_size,
                              hipStream_t stream) {
  (void)in_sizes; (void)n_in; (void)out_size; (void)ws_size;
  const float* e_v    = (const float*)d_in[0];
  const float* e_at   = (const float*)d_in[1];
  const float* e_va   = (const float*)d_in[2];
  const float* mask   = (const float*)d_in[3];
  const float* qvW  = (const float*)d_in[4];  const float* qvB  = (const float*)d_in[5];
  const float* kvW  = (const float*)d_in[6];  const float* kvB  = (const float*)d_in[7];
  const float* vvW  = (const float*)d_in[8];  const float* vvB  = (const float*)d_in[9];
  const float* qvaW = (const float*)d_in[10]; const float* qvaB = (const float*)d_in[11];
  const float* kvaW = (const float*)d_in[12]; const float* kvaB = (const float*)d_in[13];
  const float* vvaW = (const float*)d_in[14]; const float* vvaB = (const float*)d_in[15];
  const float* ovW  = (const float*)d_in[16]; const float* ovB  = (const float*)d_in[17];
  const float* ovaW = (const float*)d_in[18]; const float* ovaB = (const float*)d_in[19];
  const float* qaW  = (const float*)d_in[20]; const float* qaB  = (const float*)d_in[21];
  const float* kaW  = (const float*)d_in[22]; const float* kaB  = (const float*)d_in[23];
  const float* gate = (const float*)d_in[24];
  const float* lnvG = (const float*)d_in[25]; const float* lnvB = (const float*)d_in[26];
  const float* lnvaG= (const float*)d_in[27]; const float* lnvaB= (const float*)d_in[28];
  const float* f1W  = (const float*)d_in[29]; const float* f1B  = (const float*)d_in[30];
  const float* f2W  = (const float*)d_in[31]; const float* f2B  = (const float*)d_in[32];
  const float* flnG = (const float*)d_in[33]; const float* flnB = (const float*)d_in[34];
  const float* g1W  = (const float*)d_in[35]; const float* g1B  = (const float*)d_in[36];
  const float* g2W  = (const float*)d_in[37]; const float* g2B  = (const float*)d_in[38];
  const float* glnG = (const float*)d_in[39]; const float* glnB = (const float*)d_in[40];

  const size_t P  = (size_t)B_ * L_ * D_;  // 6,291,456
  const size_t Ph = P / 2;                 // one bf16 buffer, in float units
  const int    M  = B_ * L_;               // 8192
  float* f    = (float*)d_ws;
  float* G    = f;                         // 4 gains
  float* base = f + 64;
  // bf16 buffers (consumed by WMMA attention kernels)
  __bf16* qB  = (__bf16*)(base);
  __bf16* kB  = (__bf16*)(base + 1 * Ph);
  __bf16* vB  = (__bf16*)(base + 2 * Ph);
  __bf16* a0q = (__bf16*)(base + 3 * Ph);
  __bf16* a0k = (__bf16*)(base + 4 * Ph);
  __bf16* a1q = (__bf16*)(base + 5 * Ph);
  __bf16* a1k = (__bf16*)(base + 6 * Ph);
  // f32 temps; ctx/att overlap the attr bf16 buffers (dead after logits)
  float* ctxF = base + 3 * Ph;             // P floats (over a0q,a0k)
  float* att  = base + 5 * Ph;             // P floats (over a1q,a1k)
  float* t2   = base + 7 * Ph;             // post-LN
  float* t3   = t2 + P;                    // ffn-out
  float* big  = t3 + P;                    // logits [B*H,L,L] f32 / FFN hidden
  float* out_v  = (float*)d_out;
  float* out_va = (float*)d_out + P;

  setup_gains<<<dim3(1), dim3(4), 0, stream>>>(gate, G);

  auto gemmF = [&](const float* A, const float* W, const float* bias, float* C,
                   int m, int n, int k, int act) {
    dim3 grid(n / GN, m / GM);
    if (act)
      gemm_bf16_wmma<1, 0><<<grid, dim3(256), 0, stream>>>(A, W, bias, (void*)C, m, n, k);
    else
      gemm_bf16_wmma<0, 0><<<grid, dim3(256), 0, stream>>>(A, W, bias, (void*)C, m, n, k);
  };
  auto gemmB = [&](const float* A, const float* W, const float* bias, __bf16* C,
                   int m, int n, int k) {
    dim3 grid(n / GN, m / GM);
    gemm_bf16_wmma<0, 1><<<grid, dim3(256), 0, stream>>>(A, W, bias, (void*)C, m, n, k);
  };

  // ---------------- branch v ----------------
  gemmB(e_v, qvW, qvB, qB, M, D_, D_);
  gemmB(e_v, kvW, kvB, kB, M, D_, D_);
  gemmB(e_v, vvW, vvB, vB, M, D_, D_);
  gemmB(e_at + 0 * P, qaW + 0 * (size_t)D_ * D_, qaB + 0 * D_, a0q, M, D_, D_);
  gemmB(e_at + 0 * P, kaW + 0 * (size_t)D_ * D_, kaB + 0 * D_, a0k, M, D_, D_);
  gemmB(e_at + 1 * P, qaW + 1 * (size_t)D_ * D_, qaB + 1 * D_, a1q, M, D_, D_);
  gemmB(e_at + 1 * P, kaW + 1 * (size_t)D_ * D_, kaB + 1 * D_, a1k, M, D_, D_);
  attn_logits_wmma<<<dim3(L_ / AN, L_ / AM, B_ * H_), dim3(256), 0, stream>>>(
      qB, kB, a0q, a0k, a1q, a1k, G, 3, big);
  softmax_rows<<<dim3((B_ * H_ * L_) / 8), dim3(256), 0, stream>>>(big, mask, SCALE_);
  attn_ctx_wmma<<<dim3(1, L_ / AM, B_ * H_), dim3(256), 0, stream>>>(big, vB, ctxF);
  gemmF(ctxF, ovW, ovB, att, M, D_, D_, 0);
  add_ln_rows<<<dim3((B_ * L_) / 8), dim3(256), 0, stream>>>(att, e_v, lnvG, lnvB, t2);
  gemmF(t2, f1W, f1B, big, M, I_, D_, 1);
  gemmF(big, f2W, f2B, t3, M, D_, I_, 0);
  add_ln_rows<<<dim3((B_ * L_) / 8), dim3(256), 0, stream>>>(t3, t2, flnG, flnB, out_v);

  // ---------------- branch va ----------------
  gemmB(e_va, qvaW, qvaB, qB, M, D_, D_);
  gemmB(e_va, kvaW, kvaB, kB, M, D_, D_);
  gemmB(e_va, vvaW, vvaB, vB, M, D_, D_);
  attn_logits_wmma<<<dim3(L_ / AN, L_ / AM, B_ * H_), dim3(256), 0, stream>>>(
      qB, kB, qB, kB, qB, kB, G + 3, 1, big);
  softmax_rows<<<dim3((B_ * H_ * L_) / 8), dim3(256), 0, stream>>>(big, mask, SCALE_);
  attn_ctx_wmma<<<dim3(1, L_ / AM, B_ * H_), dim3(256), 0, stream>>>(big, vB, ctxF);
  gemmF(ctxF, ovaW, ovaB, att, M, D_, D_, 0);
  add_ln_rows<<<dim3((B_ * L_) / 8), dim3(256), 0, stream>>>(att, e_va, lnvaG, lnvaB, t2);
  gemmF(t2, g1W, g1B, big, M, I_, D_, 1);
  gemmF(big, g2W, g2B, t3, M, D_, I_, 0);
  add_ln_rows<<<dim3((B_ * L_) / 8), dim3(256), 0, stream>>>(t3, t2, glnG, glnB, out_va);
}
